// SelfAttHeadLevel1_58317065945097
// MI455X (gfx1250) — matrix operand order repaired and checked by reading, not compiled, not run on hardware
//
#include <hip/hip_runtime.h>

#ifndef __has_builtin
#define __has_builtin(x) 0
#endif

#if __has_builtin(__builtin_amdgcn_tensor_load_to_lds) && \
    __has_builtin(__builtin_amdgcn_s_wait_tensorcnt)
#define HAVE_TDM 1
#else
#define HAVE_TDM 0
#endif

#if __has_builtin(__builtin_amdgcn_cvt_pk_bf16_f32)
#define HAVE_CVT_PK 1
#else
#define HAVE_CVT_PK 0
#endif

#define B_   4
#define S_   4096
#define EMB_ 1024
#define DK_  64

typedef __attribute__((ext_vector_type(16))) __bf16 v16bf;
typedef __attribute__((ext_vector_type(8)))  float  v8f;
typedef __attribute__((ext_vector_type(4)))  unsigned int v4u;
typedef __attribute__((ext_vector_type(8)))  int v8i;
typedef __attribute__((ext_vector_type(4)))  int v4i;

union BFrag { v16bf v; unsigned short u[16]; v4u q[2]; };

__device__ __forceinline__ unsigned short f2bf(float f) {
  union { float f; unsigned u; } x; x.f = f;
  unsigned r = x.u + 0x7FFFu + ((x.u >> 16) & 1u);   // round-to-nearest-even
  return (unsigned short)(r >> 16);
}

__device__ __forceinline__ v8f wmma_bf16(const BFrag& a, const BFrag& b, v8f c) {
  return __builtin_amdgcn_wmma_f32_16x16x32_bf16(false, a.v, false, b.v,
                                                 (short)0, c, false, false);
}

// Low 32 bits of a generic pointer into shared memory == LDS byte offset
// (flat LDS aperture).
__device__ __forceinline__ unsigned lds_offset(const void* p) {
  return (unsigned)(unsigned long long)p;
}

#if HAVE_TDM
// Issue one TDM 2D tile load: 32 rows x 128 cols of bf16, row stride EMB_
// elements, packed row-major (stride 128) into LDS at ldsByteAddr.
__device__ __forceinline__ void tdm_load_v_tile(const unsigned short* gsrc,
                                                unsigned ldsByteAddr) {
  const unsigned long long ga = (unsigned long long)(size_t)gsrc;
  v4u g0;
  g0.x = 1u;                                   // count=1 (valid user D#)
  g0.y = ldsByteAddr;                          // lds_addr [63:32]
  g0.z = (unsigned)(ga & 0xFFFFFFFFull);       // global_addr[95:64]
  g0.w = (unsigned)(ga >> 32) | 0x80000000u;   // global_addr hi | type=2
  v8i g1;
  g1[0] = 0x00010000;            // workgroup_mask=0, data_size=1 (2B), no flags
  g1[1] = (int)(1024u << 16);    // atomic_barrier_addr=0, tensor_dim0[15:0]=1024
  g1[2] = (int)(4096u << 16);    // tensor_dim0[31:16]=0, tensor_dim1[15:0]=4096
  g1[3] = (int)(128u  << 16);    // tensor_dim1[31:16]=0, tile_dim0=128
  g1[4] = 32;                    // tile_dim1=32, tile_dim2=0
  g1[5] = 1024;                  // tensor_dim0_stride[31:0]=1024 elements
  g1[6] = 0;                     // stride0[47:32]=0, dim1_stride[15:0]=0
  g1[7] = 0;
  v4i z4 = {0, 0, 0, 0};
#if defined(__clang_major__) && __clang_major__ >= 23
  v8i z8 = {0, 0, 0, 0, 0, 0, 0, 0};
  __builtin_amdgcn_tensor_load_to_lds(g0, g1, z4, z4, z8, 0);
#else
  __builtin_amdgcn_tensor_load_to_lds(g0, g1, z4, z4, 0);
#endif
}
#endif

// ---------------------------------------------------------------------------
// Kernel 0: elementwise fp32 -> bf16 (done ONCE per tensor).
// ---------------------------------------------------------------------------
__global__ void cvt_f32_to_bf16_kernel(const float* __restrict__ src,
                                       unsigned short* __restrict__ dst,
                                       int n4) {
  const int i = blockIdx.x * blockDim.x + threadIdx.x;
  if (i >= n4) return;
  const float4 f = ((const float4*)src)[i];
#if HAVE_CVT_PK
  auto p0 = __builtin_amdgcn_cvt_pk_bf16_f32(f.x, f.y);
  auto p1 = __builtin_amdgcn_cvt_pk_bf16_f32(f.z, f.w);
  uint2 o;
  o.x = __builtin_bit_cast(unsigned, p0);
  o.y = __builtin_bit_cast(unsigned, p1);
  ((uint2*)dst)[i] = o;
#else
  ushort4 o;
  o.x = f2bf(f.x); o.y = f2bf(f.y); o.z = f2bf(f.z); o.w = f2bf(f.w);
  ((ushort4*)dst)[i] = o;
#endif
}

// ---------------------------------------------------------------------------
// Kernel 1: C_bf16[M,N] = bf16( (Ab[M,1024] @ Wb[N,1024]^T + bias) * postScale )
// All-bf16 inputs, pure b128 fragment loads. One wave computes 16x64
// (4 n-tiles): A-fragment reused across 4 chained WMMAs per k-step.
// postScale folds 1/sqrt(dk) into Q at projection time.
// ---------------------------------------------------------------------------
__global__ void qkv_proj_bf16_kernel(const unsigned short* __restrict__ Ab,
                                     const unsigned short* __restrict__ Wb,
                                     const float* __restrict__ bias,
                                     unsigned short* __restrict__ outBF,
                                     int N, float postScale) {
  const int lane = threadIdx.x & 31;
  const int wid  = blockIdx.x * 4 + (threadIdx.x >> 5);
  const int nblk = N >> 6;                        // 64-col slabs
  const int m0 = (wid / nblk) << 4;
  const int n0 = (wid % nblk) << 6;
  const int half = lane >> 4;
  const int lrow = lane & 15;

  v8f acc[4];
#pragma unroll
  for (int j = 0; j < 4; ++j) acc[j] = (v8f){0.f,0.f,0.f,0.f,0.f,0.f,0.f,0.f};

  const unsigned short* arow = Ab + (size_t)(m0 + lrow) * EMB_;
  const unsigned short* wrow[4];
#pragma unroll
  for (int j = 0; j < 4; ++j)
    wrow[j] = Wb + (size_t)(n0 + 16 * j + lrow) * EMB_;

  for (int k = 0; k < EMB_; k += 32) {
    BFrag af;
    const int ka = k + (half ? 16 : 0);
    af.q[0] = *(const v4u*)(arow + ka);
    af.q[1] = *(const v4u*)(arow + ka + 8);
    const int kb = k + (half ? 16 : 0);
#pragma unroll
    for (int j = 0; j < 4; ++j) {
      BFrag bfr;
      bfr.q[0] = *(const v4u*)(wrow[j] + kb);
      bfr.q[1] = *(const v4u*)(wrow[j] + kb + 8);
      acc[j] = wmma_bf16(af, bfr, acc[j]);
    }
  }

#pragma unroll
  for (int j = 0; j < 4; ++j) {
    const float bv = bias[n0 + 16 * j + lrow];
#pragma unroll
    for (int r = 0; r < 8; ++r) {
      const int row = m0 + r + (half ? 8 : 0);
      outBF[(size_t)row * N + n0 + 16 * j + lrow] =
          f2bf((acc[j][r] + bv) * postScale);
    }
  }
}

// ---------------------------------------------------------------------------
// Kernel 2: per-row softmax stats m (rowmax of scores) and l (sum exp).
// Q is pre-scaled by 1/sqrt(dk), so WMMA emits final scores directly.
// ---------------------------------------------------------------------------
__global__ void attn_stats_kernel(const unsigned short* __restrict__ Qb,
                                  const unsigned short* __restrict__ Kb,
                                  float* __restrict__ mBuf,
                                  float* __restrict__ lBuf) {
  const int lane = threadIdx.x & 31;
  const int tile = blockIdx.x * 4 + (threadIdx.x >> 5);
  const int b  = tile / (S_ / 16);
  const int q0 = (tile % (S_ / 16)) << 4;
  const int half = lane >> 4;
  const int lrow = lane & 15;

  BFrag aQ0, aQ1;
  {
    const unsigned short* qrow = Qb + ((size_t)b * S_ + q0 + lrow) * DK_;
    const int ka = half ? 16 : 0;
    aQ0.q[0] = *(const v4u*)(qrow + ka);
    aQ0.q[1] = *(const v4u*)(qrow + ka + 8);
    aQ1.q[0] = *(const v4u*)(qrow + 32 + ka);
    aQ1.q[1] = *(const v4u*)(qrow + 32 + ka + 8);
  }

  float m[8], l[8];
#pragma unroll
  for (int r = 0; r < 8; ++r) { m[r] = -1e30f; l[r] = 0.f; }

  for (int s0 = 0; s0 < S_; s0 += 16) {
    const unsigned short* krow = Kb + ((size_t)b * S_ + s0 + lrow) * DK_;
    const int kb = half ? 16 : 0;
    BFrag b0, b1;
    b0.q[0] = *(const v4u*)(krow + kb);
    b0.q[1] = *(const v4u*)(krow + kb + 8);
    b1.q[0] = *(const v4u*)(krow + 32 + kb);
    b1.q[1] = *(const v4u*)(krow + 32 + kb + 8);

    v8f sc = {0.f,0.f,0.f,0.f,0.f,0.f,0.f,0.f};
    sc = wmma_bf16(aQ0, b0, sc);
    sc = wmma_bf16(aQ1, b1, sc);

#pragma unroll
    for (int r = 0; r < 8; ++r) {
      const float v = sc[r];
      float t = v;
      t = fmaxf(t, __shfl_xor(t, 1, 32));
      t = fmaxf(t, __shfl_xor(t, 2, 32));
      t = fmaxf(t, __shfl_xor(t, 4, 32));
      t = fmaxf(t, __shfl_xor(t, 8, 32));
      const float mn = fmaxf(m[r], t);
      float e = __expf(v - mn);
      e += __shfl_xor(e, 1, 32);
      e += __shfl_xor(e, 2, 32);
      e += __shfl_xor(e, 4, 32);
      e += __shfl_xor(e, 8, 32);
      l[r] = l[r] * __expf(m[r] - mn) + e;
      m[r] = mn;
    }
  }

  if (lrow == 0) {
#pragma unroll
    for (int r = 0; r < 8; ++r) {
      const int row = q0 + r + (half ? 8 : 0);
      mBuf[b * S_ + row] = m[r];
      lBuf[b * S_ + row] = l[r];
    }
  }
}

// ---------------------------------------------------------------------------
// Kernel 3: out[16q x 128f] per block (8 waves). Waves 0-1 build the 16x32
// probability tile in LDS; the V tile (32x128 bf16 = 8KB) is DMA'd into LDS
// by the Tensor Data Mover (double-buffered, s_wait_tensorcnt) and consumed
// by all 8 waves' P@V WMMAs.
// ---------------------------------------------------------------------------
__global__ void attn_out_kernel(const unsigned short* __restrict__ Qb,
                                const unsigned short* __restrict__ Kb,
                                const unsigned short* __restrict__ Vb,
                                const float* __restrict__ mBuf,
                                const float* __restrict__ lBuf,
                                float* __restrict__ out) {
  __shared__ __align__(16) unsigned short P[16 * 32];       // bf16 prob tile
  __shared__ __align__(16) unsigned short Vt[2][32 * 128];  // staged V tiles

  const int lane = threadIdx.x & 31;
  const int wave = threadIdx.x >> 5;                        // 0..7
  const int q0 = blockIdx.x << 4;
  const int fBase = blockIdx.y << 7;
  const int fw = fBase + (wave << 4);
  const int b  = blockIdx.z;
  const int half = lane >> 4;
  const int lrow = lane & 15;

  float mrow[8], linv[8];
#pragma unroll
  for (int r = 0; r < 8; ++r) {
    const int row = q0 + r + (half ? 8 : 0);
    mrow[r] = mBuf[b * S_ + row];
    linv[r] = 1.0f / lBuf[b * S_ + row];
  }

  BFrag aQ0, aQ1;  // used by waves 0,1; loaded uniformly (valid addresses)
  {
    const unsigned short* qrow = Qb + ((size_t)b * S_ + q0 + lrow) * DK_;
    const int ka = half ? 16 : 0;
    aQ0.q[0] = *(const v4u*)(qrow + ka);
    aQ0.q[1] = *(const v4u*)(qrow + ka + 8);
    aQ1.q[0] = *(const v4u*)(qrow + 32 + ka);
    aQ1.q[1] = *(const v4u*)(qrow + 32 + ka + 8);
  }

  v8f acc = {0.f,0.f,0.f,0.f,0.f,0.f,0.f,0.f};
  const int NIT = S_ / 32;

#if HAVE_TDM
  if (wave == 7)   // prologue: DMA tile 0 into buffer 0
    tdm_load_v_tile(Vb + (size_t)(b * S_) * EMB_ + fBase, lds_offset(&Vt[0][0]));
#endif

  for (int it = 0; it < NIT; ++it) {
    const int s0 = it * 32;
#if HAVE_TDM
    const int cur = it & 1;
    if (wave == 7 && (it + 1) < NIT)
      tdm_load_v_tile(Vb + (size_t)(b * S_ + s0 + 32) * EMB_ + fBase,
                      lds_offset(&Vt[(it + 1) & 1][0]));
#else
    const int cur = 0;
    {  // cooperative synchronous staging: 512 b128 chunks, 2 per thread
      const unsigned short* gbase = Vb + (size_t)(b * S_ + s0) * EMB_ + fBase;
#pragma unroll
      for (int cc2 = 0; cc2 < 2; ++cc2) {
        const int c = threadIdx.x + cc2 * 256;      // 0..511
        const int r = c >> 4, cc = c & 15;
        *(v4u*)(&Vt[0][r * 128 + cc * 8]) =
            *(const v4u*)(gbase + (size_t)r * EMB_ + cc * 8);
      }
    }
#endif

    if (wave < 2) {  // score tile 16x16 (this wave's 16 s-columns)
      const int sc0 = s0 + (wave << 4);
      const unsigned short* krow = Kb + ((size_t)b * S_ + sc0 + lrow) * DK_;
      const int kb = half ? 16 : 0;
      BFrag b0, b1;
      b0.q[0] = *(const v4u*)(krow + kb);
      b0.q[1] = *(const v4u*)(krow + kb + 8);
      b1.q[0] = *(const v4u*)(krow + 32 + kb);
      b1.q[1] = *(const v4u*)(krow + 32 + kb + 8);
      v8f sc = {0.f,0.f,0.f,0.f,0.f,0.f,0.f,0.f};
      sc = wmma_bf16(aQ0, b0, sc);
      sc = wmma_bf16(aQ1, b1, sc);
#pragma unroll
      for (int r = 0; r < 8; ++r) {
        const float p = __expf(sc[r] - mrow[r]);   // Q pre-scaled by 1/sqrt(dk)
        P[(r + (half ? 8 : 0)) * 32 + (wave << 4) + lrow] = f2bf(p);
      }
    }

#if HAVE_TDM
    if (wave == 7) {  // ensure tile `it` is resident (in-order TDM completion)
      if ((it + 1) < NIT) __builtin_amdgcn_s_wait_tensorcnt(1);
      else                __builtin_amdgcn_s_wait_tensorcnt(0);
    }
#endif
    __syncthreads();   // publishes P and staged V tile to all waves

    // A-fragment of P from LDS (two b128 reads)
    BFrag aP;
    {
      const v4u* P4 = (const v4u*)P;
      const int base = lrow * 4;
      aP.q[0] = P4[base + 2 * half];
      aP.q[1] = P4[base + 2 * half + 1];
    }

    // B-fragment of V from LDS (16 conflict-free u16 column gathers)
    BFrag bV;
    {
      const unsigned short* vrow =
          &Vt[cur][(half ? 16 : 0) * 128 + (wave << 4) + lrow];
#pragma unroll
      for (int e = 0; e < 16; ++e) bV.u[e] = vrow[e * 128];
    }

    acc = wmma_bf16(aP, bV, acc);
    __syncthreads();   // protect P / V buffers before next iteration
  }

#pragma unroll
  for (int r = 0; r < 8; ++r) {
    const int row = q0 + r + (half ? 8 : 0);
    out[((size_t)b * S_ + row) * EMB_ + fw + lrow] = acc[r] * linv[r];
  }
}

// ---------------------------------------------------------------------------
extern "C" void kernel_launch(void* const* d_in, const int* in_sizes, int n_in,
                              void* d_out, int out_size, void* d_ws, size_t ws_size,
                              hipStream_t stream) {
  (void)in_sizes; (void)n_in; (void)out_size; (void)ws_size;
  const float* E   = (const float*)d_in[0];
  const float* WQw = (const float*)d_in[1];
  const float* WQb = (const float*)d_in[2];
  const float* WKw = (const float*)d_in[3];
  const float* WKb = (const float*)d_in[4];
  const float* WVw = (const float*)d_in[5];
  const float* WVb = (const float*)d_in[6];
  float* out = (float*)d_out;

  unsigned short* Eb   = (unsigned short*)d_ws;             // [B*S,1024] bf16
  unsigned short* WQbw = Eb   + (size_t)B_ * S_ * EMB_;     // [64,1024]
  unsigned short* WKbw = WQbw + (size_t)DK_ * EMB_;         // [64,1024]
  unsigned short* WVbw = WKbw + (size_t)DK_ * EMB_;         // [1024,1024]
  unsigned short* Qb   = WVbw + (size_t)EMB_ * EMB_;        // [B*S,64]
  unsigned short* Kb   = Qb   + (size_t)B_ * S_ * DK_;      // [B*S,64]
  unsigned short* Vb   = Kb   + (size_t)B_ * S_ * DK_;      // [B*S,1024]
  float* mBuf = (float*)(Vb + (size_t)B_ * S_ * EMB_);      // [B*S]
  float* lBuf = mBuf + (size_t)B_ * S_;                     // [B*S]

  const int M = B_ * S_;  // 16384

  // 0) one-time fp32 -> bf16 conversions
  {
    int n4 = M * EMB_ / 4;                                  // 4,194,304
    cvt_f32_to_bf16_kernel<<<n4 / 256, 256, 0, stream>>>(E, Eb, n4);
    n4 = DK_ * EMB_ / 4;                                    // 16,384
    cvt_f32_to_bf16_kernel<<<n4 / 256, 256, 0, stream>>>(WQw, WQbw, n4);
    cvt_f32_to_bf16_kernel<<<n4 / 256, 256, 0, stream>>>(WKw, WKbw, n4);
    n4 = EMB_ * EMB_ / 4;                                   // 262,144
    cvt_f32_to_bf16_kernel<<<n4 / 256, 256, 0, stream>>>(WVw, WVbw, n4);
  }

  // 1) QKV projections (bf16 WMMA GEMMs); Q carries the 1/sqrt(dk) scale
  {
    const int wavesQK = (M / 16) * (DK_ / 64);              // 1024
    qkv_proj_bf16_kernel<<<wavesQK / 4, 128, 0, stream>>>(Eb, WQbw, WQb, Qb, DK_, 0.125f);
    qkv_proj_bf16_kernel<<<wavesQK / 4, 128, 0, stream>>>(Eb, WKbw, WKb, Kb, DK_, 1.0f);
    const int wavesV = (M / 16) * (EMB_ / 64);              // 16384
    qkv_proj_bf16_kernel<<<wavesV / 4, 128, 0, stream>>>(Eb, WVbw, WVb, Vb, EMB_, 1.0f);
  }

  // 2) softmax statistics
  attn_stats_kernel<<<(B_ * (S_ / 16)) / 4, 128, 0, stream>>>(Qb, Kb, mBuf, lBuf);

  // 3) attention output
  {
    dim3 grid(S_ / 16, EMB_ / 128, B_);                     // (256, 8, 4)
    attn_out_kernel<<<grid, 256, 0, stream>>>(Qb, Kb, Vb, mBuf, lBuf, out);
  }
}
